// Encoder_77455440216069
// MI455X (gfx1250) — compile-verified, matrix operand order checked by reading, not run on hardware
//
#include <hip/hip_runtime.h>

// ---------------------------------------------------------------------------
// CDNA5 / gfx1250 graph-transformer encoder.
// bf16 WMMA (v_wmma_f32_16x16x32_bf16), f32 accumulate.
// Layouts chosen so every WMMA operand fragment is 1-2 contiguous 16B runs
// per lane:  Q/K bf16 [feat][head] (col=a*32+h),  V bf16 [head][dv]
// (col=h*16+dv),  agg f32 [head*16+dv] (reference combine order).
// Softmax / LN reductions via ds_swizzle_b32 xor mode (no bpermute overhead).
// ---------------------------------------------------------------------------

typedef __attribute__((ext_vector_type(16))) __bf16          v16bf;
typedef __attribute__((ext_vector_type(16))) unsigned short  v16u;
typedef __attribute__((ext_vector_type(8)))  unsigned short  v8u;
typedef __attribute__((ext_vector_type(8)))  float           v8f;
typedef __attribute__((ext_vector_type(4)))  float           f4;

static __device__ __forceinline__ unsigned short f2b(float f) {
  unsigned int u = __float_as_uint(f);
  u += 0x7FFFu + ((u >> 16) & 1u);          // round-to-nearest-even
  return (unsigned short)(u >> 16);
}

static __device__ __forceinline__ v16bf as_bf(v16u u) {
  union { v16u u; v16bf b; } c; c.u = u; return c.b;
}

static __device__ __forceinline__ v16u cat8(v8u lo, v8u hi) {
  return __builtin_shufflevector(lo, hi, 0,1,2,3,4,5,6,7,8,9,10,11,12,13,14,15);
}

// ds_swizzle_b32, group-of-32 XOR mode: offset = xor<<10 | or<<5 | and
#define XSWZ(v, m) \
  __int_as_float(__builtin_amdgcn_ds_swizzle(__float_as_int(v), (((m) << 10) | 0x1f)))

// bf16 16x32 fragment element->K map (ISA 7.12.2):
//   e in [0,8):  k = hi8 + e        (hi8 = 0 for lanes 0-15, 8 for lanes 16-31)
//   e in [8,16): k = 16 + hi8 + (e-8)
static __device__ __forceinline__ int frag_k(int e, int hi8) {
  int vg = e >> 1, pos = e & 1;
  return ((vg >> 2) << 4) + hi8 + ((vg & 3) << 1) + pos;
}

// ---------------------------------------------------------------------------
// Tiled GEMM: C[M,Nout] = A[M,K] (f32 row-major) x W[Nout,K]^T (f32 row-major)
// MODE 0: f32 out, identity cols.   MODE 1: bf16 out, identity (V).
// MODE 2: bf16 out, col (head*16+feat) stored at feat*32+head (Q/K).
// ---------------------------------------------------------------------------
template <int MODE>
__global__ void gemm_wmma(const float* __restrict__ A, const float* __restrict__ W,
                          void* __restrict__ Cout, int M, int Nout, int K)
{
  __shared__ v8u As4[16][5];                 // 16 rows x 40 ushorts (80B stride)
  unsigned short* Ass = (unsigned short*)As4;
  const int lane = threadIdx.x & 31;
  const int wave = threadIdx.x >> 5;
  const int m0   = blockIdx.x * 16;
  const int n0   = blockIdx.y * 128 + wave * 16;
  const int l15  = lane & 15;
  const int hi8  = (lane >> 4) << 3;
  const int c8   = hi8 >> 3;                 // 0 or 1
  const float* wrow = W + (size_t)(n0 + l15) * K;
  const bool vec_ok = ((K & 31) == 0);       // whole-tile, 16B-aligned fast path

  v8f acc = {0.f, 0.f, 0.f, 0.f, 0.f, 0.f, 0.f, 0.f};

  for (int k0 = 0; k0 < K; k0 += 32) {
    // cooperative A-tile (16 x 32) f32 -> bf16 into LDS
    for (int i = threadIdx.x; i < 512; i += 256) {
      int r = i >> 5, kk = i & 31;
      int gm = m0 + r, gk = k0 + kk;
      float f = (gm < M && gk < K) ? A[(size_t)gm * K + gk] : 0.f;
      Ass[r * 40 + kk] = f2b(f);
      if (gm < M && gk + 32 < K)
        __builtin_prefetch(&A[(size_t)gm * K + gk + 32], 0, 1);  // global_prefetch_b8
    }
    __syncthreads();

    // A fragment: two ds_load_b128 per lane
    v16u au = cat8(As4[l15][c8], As4[l15][2 + c8]);

    // B fragment from W
    v16u bu;
    if (vec_ok) {
      const float* wp = wrow + k0;
      f4 a0 = *(const f4*)(wp + hi8);
      f4 a1 = *(const f4*)(wp + hi8 + 4);
      f4 a2 = *(const f4*)(wp + 16 + hi8);
      f4 a3 = *(const f4*)(wp + 16 + hi8 + 4);
#pragma unroll
      for (int j = 0; j < 4; ++j) {
        bu[j]      = f2b(a0[j]);
        bu[4 + j]  = f2b(a1[j]);
        bu[8 + j]  = f2b(a2[j]);
        bu[12 + j] = f2b(a3[j]);
      }
    } else {
#pragma unroll
      for (int e = 0; e < 16; ++e) {
        int gk = k0 + frag_k(e, hi8);
        bu[e] = (gk < K) ? f2b(wrow[gk]) : (unsigned short)0;
      }
    }

    acc = __builtin_amdgcn_wmma_f32_16x16x32_bf16(
        false, as_bf(au), false, as_bf(bu), (short)0, acc, false, false);
    __syncthreads();
  }

  // D layout: VGPR r -> row m0+r+hi8, col n0+l15
  const int c = n0 + l15;
#pragma unroll
  for (int r = 0; r < 8; ++r) {
    int gm = m0 + r + hi8;
    if (gm < M) {
      if (MODE == 0)
        ((float*)Cout)[(size_t)gm * Nout + c] = acc[r];
      else if (MODE == 1)
        ((unsigned short*)Cout)[(size_t)gm * Nout + c] = f2b(acc[r]);
      else  // MODE 2: c = head*16+feat -> store at feat*32+head
        ((unsigned short*)Cout)[(size_t)gm * Nout + (c & 15) * 32 + (c >> 4)] = f2b(acc[r]);
    }
  }
}

// ---------------------------------------------------------------------------
// Edge attention: one wave per edge, 3 WMMAs/edge.
//   alpha[a][b] = sum_h Q[dst][a][h] K[src][b][h]   (one 16x16x32 WMMA)
//   softmax over b (16-lane-group ds_swizzle reductions)
//   msg[a][h]   = sum_b att[a][b] V[src][b][h]      (two WMMAs, K padded to 32)
//   atomic scatter into agg[dst], col = h*16 + a
// ---------------------------------------------------------------------------
__global__ void edge_attn(const int* __restrict__ ei,
                          const unsigned short* __restrict__ Qb,
                          const unsigned short* __restrict__ Kb,
                          const unsigned short* __restrict__ Vb,
                          float* __restrict__ agg, int E_)
{
  __shared__ v8u attV[8][32];                // per-wave 16x16 bf16 att matrix
  const int lane = threadIdx.x & 31;
  const int wave = threadIdx.x >> 5;
  const int eid  = blockIdx.x * 8 + wave;
  const bool active = eid < E_;
  const int e    = active ? eid : 0;
  const int src  = ei[e];
  const int dst  = ei[E_ + e];
  const int l15  = lane & 15;
  const int hi8  = (lane >> 4) << 3;
  const int c8   = hi8 >> 3;

  const unsigned short* q  = Qb + (size_t)dst * 512;   // [a][h]: a*32+h
  const unsigned short* kp = Kb + (size_t)src * 512;   // [b][h]: b*32+h
  const unsigned short* vp = Vb + (size_t)src * 512;   // [h][dv]: h*16+dv

  // Q (A op, m=a=l15, k=h) and K (B op, n=b=l15, k=h): 2 x b128 each
  v16u qu = cat8(*(const v8u*)(q  + l15 * 32 + hi8),
                 *(const v8u*)(q  + l15 * 32 + 16 + hi8));
  v16u ku = cat8(*(const v8u*)(kp + l15 * 32 + hi8),
                 *(const v8u*)(kp + l15 * 32 + 16 + hi8));

  v8f alpha = {0.f, 0.f, 0.f, 0.f, 0.f, 0.f, 0.f, 0.f};
  alpha = __builtin_amdgcn_wmma_f32_16x16x32_bf16(
      false, as_bf(qu), false, as_bf(ku), (short)0, alpha, false, false);

  // softmax over b: row a fixed within a 16-lane group, col b = l15
  float att[8];
#pragma unroll
  for (int r = 0; r < 8; ++r) {
    float v  = alpha[r] * 0.25f;             // 1/sqrt(DK)
    float mx = v;
    mx = fmaxf(mx, XSWZ(mx, 8));
    mx = fmaxf(mx, XSWZ(mx, 4));
    mx = fmaxf(mx, XSWZ(mx, 2));
    mx = fmaxf(mx, XSWZ(mx, 1));
    float ex = __expf(v - mx);
    float sm = ex;
    sm += XSWZ(sm, 8);
    sm += XSWZ(sm, 4);
    sm += XSWZ(sm, 2);
    sm += XSWZ(sm, 1);
    att[r] = ex / sm;
  }

  // redistribute att (C/D layout) -> A-operand layout via LDS
  unsigned short* ats = (unsigned short*)attV[wave];
#pragma unroll
  for (int r = 0; r < 8; ++r)
    ats[(r + hi8) * 16 + l15] = f2b(att[r]);
  __syncthreads();

  const v8u z8 = (v8u)0;
  // att A op: m=a=l15, k=b (only k<16 valid): one ds_load_b128
  v16u au  = cat8(attV[wave][l15 * 2 + c8], z8);
  // V B ops: n=h, k=b (k<16): one b128 each
  v16u v0u = cat8(*(const v8u*)(vp + l15 * 16 + hi8), z8);          // h = l15
  v16u v1u = cat8(*(const v8u*)(vp + (16 + l15) * 16 + hi8), z8);   // h = 16+l15

  v8f z = {0.f, 0.f, 0.f, 0.f, 0.f, 0.f, 0.f, 0.f};
  v8f m0 = __builtin_amdgcn_wmma_f32_16x16x32_bf16(
      false, as_bf(au), false, as_bf(v0u), (short)0, z, false, false);
  v8f m1 = __builtin_amdgcn_wmma_f32_16x16x32_bf16(
      false, as_bf(au), false, as_bf(v1u), (short)0, z, false, false);

  if (active) {
    float* ap = agg + (size_t)dst * 512;
#pragma unroll
    for (int r = 0; r < 8; ++r) {
      int a = r + hi8;
      atomicAdd(&ap[l15 * 16 + a],        m0[r]);   // h = l15
      atomicAdd(&ap[(16 + l15) * 16 + a], m1[r]);   // h = 16+l15
    }
  }
}

// ---------------------------------------------------------------------------
// out = LayerNorm(base + delta + bias) * g + b    (D = 256, one wave per row)
// ---------------------------------------------------------------------------
__global__ void resid_ln(const float* __restrict__ base, const float* __restrict__ delta,
                         const float* __restrict__ bias, const float* __restrict__ g,
                         const float* __restrict__ b, float* __restrict__ out, int nrows)
{
  const int lane = threadIdx.x & 31;
  const int row  = blockIdx.x * 8 + (threadIdx.x >> 5);
  if (row >= nrows) return;
  const float* bp = base  + (size_t)row * 256;
  const float* dp = delta + (size_t)row * 256;
  float v[8], s = 0.f;
#pragma unroll
  for (int j = 0; j < 8; ++j) {
    int c = j * 32 + lane;
    v[j] = bp[c] + dp[c] + bias[c];
    s += v[j];
  }
  s += XSWZ(s, 16); s += XSWZ(s, 8); s += XSWZ(s, 4); s += XSWZ(s, 2); s += XSWZ(s, 1);
  float mean = s * (1.f / 256.f);
  float s2 = 0.f;
#pragma unroll
  for (int j = 0; j < 8; ++j) { float d = v[j] - mean; s2 += d * d; }
  s2 += XSWZ(s2, 16); s2 += XSWZ(s2, 8); s2 += XSWZ(s2, 4); s2 += XSWZ(s2, 2); s2 += XSWZ(s2, 1);
  float inv = rsqrtf(s2 * (1.f / 256.f) + 1e-5f);
#pragma unroll
  for (int j = 0; j < 8; ++j) {
    int c = j * 32 + lane;
    out[(size_t)row * 256 + c] = (v[j] - mean) * inv * g[c] + b[c];
  }
}

__global__ void zero_f32(float* __restrict__ p, size_t n) {
  size_t i = (size_t)blockIdx.x * blockDim.x + threadIdx.x;
  size_t stride = (size_t)gridDim.x * blockDim.x;
  for (; i < n; i += stride) p[i] = 0.f;
}

// ---------------------------------------------------------------------------
extern "C" void kernel_launch(void* const* d_in, const int* in_sizes, int n_in,
                              void* d_out, int out_size, void* d_ws, size_t ws_size,
                              hipStream_t stream)
{
  const float* x     = (const float*)d_in[0];
  const int*   ei    = (const int*)  d_in[1];
  const float* Wemb  = (const float*)d_in[2];
  const float* Wq    = (const float*)d_in[3];
  const float* Wk    = (const float*)d_in[4];
  const float* Wv    = (const float*)d_in[5];
  const float* Wo    = (const float*)d_in[6];
  const float* bo    = (const float*)d_in[7];
  const float* Wm    = (const float*)d_in[8];
  const float* bm    = (const float*)d_in[9];
  const float* g_ln  = (const float*)d_in[10];
  const float* b_ln  = (const float*)d_in[11];
  const float* g_mlp = (const float*)d_in[12];
  const float* b_mlp = (const float*)d_in[13];
  (void)n_in; (void)out_size; (void)ws_size;

  const int Dm  = 256;
  const int INf = in_sizes[2] / Dm;                 // 3703
  const int Nn  = in_sizes[0] / INf;                // 10000
  const int Ee  = in_sizes[1] / 2;                  // 160000
  const int Ll  = in_sizes[3] / (512 * Dm);         // 7

  char* wsp = (char*)d_ws;
  auto carve = [&](size_t bytes) {
    void* p = (void*)wsp;
    wsp += (bytes + 255) & ~(size_t)255;
    return p;
  };
  unsigned short* Qb  = (unsigned short*)carve((size_t)Nn * 512 * 2);
  unsigned short* Kb  = (unsigned short*)carve((size_t)Nn * 512 * 2);
  unsigned short* Vb  = (unsigned short*)carve((size_t)Nn * 512 * 2);
  float*          agg = (float*)carve((size_t)Nn * 512 * 4);
  float*          t   = (float*)carve((size_t)Nn * 256 * 4);
  float*          h   = (float*)carve((size_t)Nn * 256 * 4);
  float*          h1  = (float*)carve((size_t)Nn * 256 * 4);

  const int mtiles = (Nn + 15) / 16;
  dim3 blk(256);

  // embed: h = x @ W_embed^T
  gemm_wmma<0><<<dim3(mtiles, 256 / 128), blk, 0, stream>>>(x, Wemb, h, Nn, 256, INf);

  for (int l = 0; l < Ll; ++l) {
    const float* Wq_l = Wq + (size_t)l * 512 * Dm;
    const float* Wk_l = Wk + (size_t)l * 512 * Dm;
    const float* Wv_l = Wv + (size_t)l * 512 * Dm;
    const float* Wo_l = Wo + (size_t)l * Dm * 512;
    const float* Wm_l = Wm + (size_t)l * Dm * Dm;
    const float* bo_l = bo + (size_t)l * Dm;
    const float* bm_l = bm + (size_t)l * Dm;
    const float* gl_l = g_ln  + (size_t)l * Dm;
    const float* bl_l = b_ln  + (size_t)l * Dm;
    const float* gm_l = g_mlp + (size_t)l * Dm;
    const float* bmp  = b_mlp + (size_t)l * Dm;

    dim3 gq(mtiles, 512 / 128);
    gemm_wmma<2><<<gq, blk, 0, stream>>>(h, Wq_l, Qb, Nn, 512, Dm);  // [feat][head]
    gemm_wmma<2><<<gq, blk, 0, stream>>>(h, Wk_l, Kb, Nn, 512, Dm);  // [feat][head]
    gemm_wmma<1><<<gq, blk, 0, stream>>>(h, Wv_l, Vb, Nn, 512, Dm);  // [head][dv]

    zero_f32<<<2048, blk, 0, stream>>>(agg, (size_t)Nn * 512);
    edge_attn<<<(Ee + 7) / 8, blk, 0, stream>>>(ei, Qb, Kb, Vb, agg, Ee);

    dim3 gp(mtiles, 256 / 128);
    gemm_wmma<0><<<gp, blk, 0, stream>>>(agg, Wo_l, t, Nn, 256, 512);
    resid_ln<<<(Nn + 7) / 8, blk, 0, stream>>>(h, t, bo_l, gl_l, bl_l, h1, Nn);

    gemm_wmma<0><<<gp, blk, 0, stream>>>(h1, Wm_l, t, Nn, 256, Dm);
    float* outp = (l == Ll - 1) ? (float*)d_out : h;
    resid_ln<<<(Nn + 7) / 8, blk, 0, stream>>>(h1, t, bm_l, gm_l, bmp, outp, Nn);
  }
}
